// ChannelAttention_14181982011826
// MI455X (gfx1250) — matrix operand order repaired and checked by reading, not compiled, not run on hardware
//
#include <hip/hip_runtime.h>
#include <cstdint>

typedef __bf16 bf16;
typedef __attribute__((ext_vector_type(16))) __bf16 v16bf;
typedef __attribute__((ext_vector_type(8)))  float  v8f;
typedef uint32_t u32x4 __attribute__((ext_vector_type(4)));
typedef int      i32x4 __attribute__((ext_vector_type(4)));
typedef int      i32x8 __attribute__((ext_vector_type(8)));

// ---------------------------------------------------------------- helpers ----

__device__ __forceinline__ bf16 f2bf(float f) {
  union { float f; uint32_t u; } v; v.f = f;
  uint32_t r = (v.u + 0x7FFFu + ((v.u >> 16) & 1u)) >> 16;  // round-nearest-even
  uint16_t s = (uint16_t)r;
  return __builtin_bit_cast(bf16, s);
}

__device__ __forceinline__ v8f zero8() {
  v8f z;
#pragma unroll
  for (int i = 0; i < 8; ++i) z[i] = 0.0f;
  return z;
}

// A fragment: 16x32 bf16 tile at p (row-major, row stride = pitch elements).
// CDNA5 layout: lanes 0-15 -> M=lane, K in {0..7,16..23}; lanes 16-31 -> K in {8..15,24..31}.
// Each half is 8 contiguous elements -> two b128 loads per fragment.
__device__ __forceinline__ v16bf load_afrag(const bf16* p, int pitch, int lane) {
  const int h = lane >> 4, m = lane & 15;
  v16bf a;
#pragma unroll
  for (int j = 0; j < 8; ++j) {
    const int kk = 8 * h + (j < 4 ? 2 * j : 16 + 2 * (j - 4));
    a[2 * j]     = p[m * pitch + kk];
    a[2 * j + 1] = p[m * pitch + kk + 1];
  }
  return a;
}

// B fragment from a TRANSPOSED tile Bt[col][k] (row stride = pitch elements).
// Lane layout: lanes 0-15 -> col=lane, K in {0..7,16..23}; lanes 16-31 -> col=lane-16, K in {8..15,24..31}.
// Per lane this is two runs of 8 contiguous bf16 -> two b128 loads per fragment.
__device__ __forceinline__ v16bf load_bfragT(const bf16* p, int pitch, int lane) {
  const int h = lane >> 4, n = lane & 15;
  v16bf b;
#pragma unroll
  for (int e = 0; e < 16; ++e) b[e] = p[n * pitch + 8 * h + (e < 8 ? e : 16 + (e - 8))];
  return b;
}

__device__ __forceinline__ v8f wmma_bf16(v16bf a, v16bf b, v8f c) {
  return __builtin_amdgcn_wmma_f32_16x16x32_bf16(false, a, false, b, (short)0, c,
                                                 false, false);
}

// ------------------------------------------------ Tensor Data Mover (TDM) ----
// 2D tile DMA: global (row-major, stride_d0 elems between rows, 2-byte elems)
// -> LDS at lds_byte_addr with hardware row padding (pitch = tile_d0 + pad).
//   pad_int_code: DWORDs/row before pad (3=16DW,4=32DW,5=64DW,7=256DW)
//   pad_amt_code: pad DWORDs - 1

__device__ __forceinline__ uint32_t lds_off(const void* p) {
  return (uint32_t)(uintptr_t)p;   // low 32 bits of flat shared addr = LDS byte offset
}

__device__ __forceinline__ void tdm_load_2d(uint32_t lds_byte_addr, const void* gptr,
                                            uint32_t tile_d0, uint32_t tile_d1,
                                            uint32_t stride_d0,
                                            uint32_t pad_int_code,
                                            uint32_t pad_amt_code) {
  const uint64_t ga = (uint64_t)(uintptr_t)gptr;
  u32x4 g0;
  g0[0] = 1u;                                  // count=1: one valid user D#
  g0[1] = lds_byte_addr;                       // lds_addr
  g0[2] = (uint32_t)ga;                        // global_addr[31:0]
  g0[3] = (uint32_t)(ga >> 32) | (2u << 30);   // global_addr[56:32] | type=2(image)
  i32x8 g1;
  g1[0] = (int)((1u << 16) |                   // data_size=1 (2 bytes)
                (1u << 20) |                   // pad_enable
                (pad_int_code << 22) | (pad_amt_code << 25));
  g1[1] = (int)((tile_d0 & 0xFFFFu) << 16);                 // tensor_dim0 lo (=tile_d0)
  g1[2] = (int)(((tile_d0 >> 16) & 0xFFFFu) |               // tensor_dim0 hi
                ((tile_d1 & 0xFFFFu) << 16));               // tensor_dim1 lo
  g1[3] = (int)(((tile_d1 >> 16) & 0xFFFFu) |               // tensor_dim1 hi
                ((tile_d0 & 0xFFFFu) << 16));               // tile_dim0
  g1[4] = (int)(tile_d1 & 0xFFFFu);                         // tile_dim1 (tile_dim2=0)
  g1[5] = (int)stride_d0;                                   // tensor_dim0_stride lo
  g1[6] = 0;
  g1[7] = 0;
  const i32x4 gz = {0, 0, 0, 0};
#if defined(__clang_major__) && (__clang_major__ >= 23)
  const i32x8 gz8 = {0, 0, 0, 0, 0, 0, 0, 0};
  __builtin_amdgcn_tensor_load_to_lds(g0, g1, gz, gz, gz8, 0);
#else
  __builtin_amdgcn_tensor_load_to_lds(g0, g1, gz, gz, 0);
#endif
}

// ----------------------------------------------------------- K0: fp32->bf16 --

__global__ __launch_bounds__(256) void cvt_f32_bf16(const float* __restrict__ src,
                                                    bf16* __restrict__ dst, int n4) {
  const int i = blockIdx.x * 256 + threadIdx.x;
  if (i < n4) {
    const float4 f = ((const float4*)src)[i];
    const uint32_t lo = ((uint32_t)__builtin_bit_cast(uint16_t, f2bf(f.y)) << 16) |
                        (uint32_t)__builtin_bit_cast(uint16_t, f2bf(f.x));
    const uint32_t hi = ((uint32_t)__builtin_bit_cast(uint16_t, f2bf(f.w)) << 16) |
                        (uint32_t)__builtin_bit_cast(uint16_t, f2bf(f.z));
    ((uint2*)dst)[i] = make_uint2(lo, hi);
  }
}

// --------------------------- K0b: fp32 [R][C] -> bf16 transposed [C][R] ------

__global__ __launch_bounds__(256) void cvt_f32_bf16_t(const float* __restrict__ src,
                                                      bf16* __restrict__ dst,
                                                      int R, int C) {
  __shared__ bf16 t[32][33];
  const int tx = threadIdx.x & 31, ty = threadIdx.x >> 5;
#pragma unroll
  for (int i = 0; i < 4; ++i) {
    const int r = blockIdx.y * 32 + ty + i * 8;
    const int c = blockIdx.x * 32 + tx;
    t[ty + i * 8][tx] = f2bf(src[(size_t)r * C + c]);
  }
  __syncthreads();
#pragma unroll
  for (int i = 0; i < 4; ++i) {
    const int orow = blockIdx.x * 32 + ty + i * 8;   // original col
    const int ocol = blockIdx.y * 32 + tx;           // original row
    dst[(size_t)orow * R + ocol] = t[tx][ty + i * 8];
  }
}

// -------------------------------------------------- K1: qkv = x @ Wqkv -------
// M=32768, Ncol=3072, K=1024.  Block tile 128x128, KT=32, 8 waves (4x2),
// wave tile 32x64 = 2x4 WMMA tiles.  A from x(bf16), B from WqkvT (both TDM).

__global__ __launch_bounds__(256) void qkv_gemm_kernel(
    const bf16* __restrict__ xb, const bf16* __restrict__ wqT,
    bf16* __restrict__ qb, bf16* __restrict__ kb, bf16* __restrict__ vb) {
  __shared__ __align__(16) bf16 As[128 * 40];   // 128 rows(m) x 32(k), pitch 40
  __shared__ __align__(16) bf16 Bs[128 * 40];   // 128 rows(col) x 32(k), pitch 40
  const int tid  = threadIdx.x;
  const int lane = tid & 31;
  const int wave = tid >> 5;
  const int bm = blockIdx.x * 128;
  const int bn = blockIdx.y * 128;
  const int wm = (wave >> 1) * 32;
  const int wn = (wave & 1) * 64;

  v8f acc[2][4];
#pragma unroll
  for (int i = 0; i < 2; ++i)
#pragma unroll
    for (int j = 0; j < 4; ++j) acc[i][j] = zero8();

  for (int k0 = 0; k0 < 1024; k0 += 32) {
    if (wave == 0) {
      tdm_load_2d(lds_off(As), xb  + (size_t)bm * 1024 + k0, 32, 128, 1024, 3, 3);
      tdm_load_2d(lds_off(Bs), wqT + (size_t)bn * 1024 + k0, 32, 128, 1024, 3, 3);
      __builtin_amdgcn_s_wait_tensorcnt(0);
    }
    __syncthreads();
    v16bf af[2], bfr[4];
#pragma unroll
    for (int mt = 0; mt < 2; ++mt) af[mt] = load_afrag(As + (wm + mt * 16) * 40, 40, lane);
#pragma unroll
    for (int nt = 0; nt < 4; ++nt) bfr[nt] = load_bfragT(Bs + (wn + nt * 16) * 40, 40, lane);
#pragma unroll
    for (int mt = 0; mt < 2; ++mt)
#pragma unroll
      for (int nt = 0; nt < 4; ++nt)
        acc[mt][nt] = wmma_bf16(af[mt], bfr[nt], acc[mt][nt]);
    __syncthreads();
  }

  const int h = lane >> 4, nn = lane & 15;
#pragma unroll
  for (int mt = 0; mt < 2; ++mt)
#pragma unroll
    for (int nt = 0; nt < 4; ++nt)
#pragma unroll
      for (int r = 0; r < 8; ++r) {
        const int grow = bm + wm + mt * 16 + r + 8 * h;   // b*4096 + n
        const int gcol = bn + wn + nt * 16 + nn;          // 0..3071
        const float val = acc[mt][nt][r];
        const int which = gcol >> 10, cc = gcol & 1023;
        const int g = cc >> 6, d = cc & 63;
        const int b = grow >> 12, nidx = grow & 4095;
        const size_t bg = (size_t)(b * 16 + g);
        if (which == 0)      qb[(bg * 4096 + nidx) * 64 + d] = f2bf(val);          // q[n][d]
        else if (which == 1) kb[(bg * 64 + d) * 4096 + nidx] = f2bf(val * 0.125f); // kT, *hd^-0.5
        else                 vb[(bg * 64 + d) * 4096 + nidx] = f2bf(val);          // vT
      }
}

// --------------------------- K2: channel attention (per b,g) -----------------
// pass1: attn[d,e] = sum_n kT[d,n] * vT[e,n]   (M=N=64, K=4096, bf16 WMMA)
// softmax rows (fp32, LDS) -> attnS[d][e] bf16 (already "transposed-B" layout),
// pass2: out[n,d] = sum_e q[n,e] * attn[d,e].

__global__ __launch_bounds__(256) void chan_attn_kernel(
    const bf16* __restrict__ qb, const bf16* __restrict__ kb,
    const bf16* __restrict__ vb, bf16* __restrict__ ao) {
  __shared__ __align__(16) bf16 As[64 * 40];      // kT tile   [d][n-chunk]
  __shared__ __align__(16) bf16 Bs[64 * 40];      // vT tile   [e][n-chunk]
  __shared__ float attn[64 * 68];
  __shared__ __align__(16) bf16 attnS[64 * 72];   // softmax(attn)[d][e], pitch 72

  const int tid = threadIdx.x, lane = tid & 31, wave = tid >> 5;
  const int bg = blockIdx.x;
  const int b = bg >> 4, g = bg & 15;
  const bf16* kmat = kb + (size_t)bg * 64 * 4096;   // [d][n]
  const bf16* vmat = vb + (size_t)bg * 64 * 4096;   // [e][n]
  const bf16* qmat = qb + (size_t)bg * 4096 * 64;   // [n][e]
  const int h = lane >> 4, nn = lane & 15;

  const int tm  = wave >> 1;           // M-tile (d) of this wave
  const int tn0 = (wave & 1) * 2;      // first of two N-tiles (e)

  v8f acc0 = zero8(), acc1 = zero8();
  for (int k0 = 0; k0 < 4096; k0 += 32) {
    if (wave == 0) {
      tdm_load_2d(lds_off(As), kmat + k0, 32, 64, 4096, 3, 3);
      tdm_load_2d(lds_off(Bs), vmat + k0, 32, 64, 4096, 3, 3);
      __builtin_amdgcn_s_wait_tensorcnt(0);
    }
    __syncthreads();
    v16bf a  = load_afrag(As + tm * 16 * 40, 40, lane);
    v16bf b0 = load_bfragT(Bs + tn0 * 16 * 40, 40, lane);
    v16bf b1 = load_bfragT(Bs + (tn0 + 1) * 16 * 40, 40, lane);
    acc0 = wmma_bf16(a, b0, acc0);
    acc1 = wmma_bf16(a, b1, acc1);
    __syncthreads();
  }
#pragma unroll
  for (int r = 0; r < 8; ++r) {
    attn[(tm * 16 + r + 8 * h) * 68 + tn0 * 16 + nn]       = acc0[r];
    attn[(tm * 16 + r + 8 * h) * 68 + (tn0 + 1) * 16 + nn] = acc1[r];
  }
  __syncthreads();
  if (tid < 64) {  // fp32 softmax, one row (d) per thread; write bf16 row-major
    float mx = -3.4e38f;
    for (int e = 0; e < 64; ++e) mx = fmaxf(mx, attn[tid * 68 + e]);
    float s = 0.f;
    for (int e = 0; e < 64; ++e) {
      const float p = __expf(attn[tid * 68 + e] - mx);
      attn[tid * 68 + e] = p; s += p;
    }
    const float inv = 1.0f / s;
    for (int e = 0; e < 64; ++e) attnS[tid * 72 + e] = f2bf(attn[tid * 68 + e] * inv);
  }
  __syncthreads();

  // attnS B fragments (cols d x K e) live in registers for the pass-2 sweep.
  v16bf bf2[2][4];
#pragma unroll
  for (int kk = 0; kk < 2; ++kk)
#pragma unroll
    for (int nt = 0; nt < 4; ++nt)
      bf2[kk][nt] = load_bfragT(attnS + nt * 16 * 72 + kk * 32, 72, lane);

  for (int mt = 0; mt < 32; ++mt) {     // each wave owns 512 rows of n
    const int m0 = wave * 512 + mt * 16;
    v16bf a0 = load_afrag(qmat + (size_t)m0 * 64, 64, lane);
    v16bf a1 = load_afrag(qmat + (size_t)m0 * 64 + 32, 64, lane);
    v8f o[4];
#pragma unroll
    for (int nt = 0; nt < 4; ++nt) {
      o[nt] = zero8();
      o[nt] = wmma_bf16(a0, bf2[0][nt], o[nt]);
      o[nt] = wmma_bf16(a1, bf2[1][nt], o[nt]);
    }
#pragma unroll
    for (int nt = 0; nt < 4; ++nt)
#pragma unroll
      for (int r = 0; r < 8; ++r) {
        const int n = m0 + r + 8 * h;
        ao[((size_t)b * 4096 + n) * 1024 + g * 64 + nt * 16 + nn] = f2bf(o[nt][r]);
      }
  }
}

// ------------------- K3: out = LN(x + ao @ Wproj + bproj) --------------------
// Block owns 16 full rows (all 1024 cols): LayerNorm reduces inside the block
// via LDS ds_add_f32 atomics.  8 waves x 64-col slices; two 512-col halves of
// WprojT staged per K-step by the TDM.

__global__ __launch_bounds__(256) void proj_ln_kernel(
    const bf16* __restrict__ ao, const bf16* __restrict__ wpT,
    const float* __restrict__ x, const float* __restrict__ bproj,
    const float* __restrict__ gamma, const float* __restrict__ beta,
    float* __restrict__ out) {
  __shared__ __align__(16) bf16 As[16 * 40];
  __shared__ __align__(16) bf16 Bs[512 * 40];     // 512 rows(col) x 32(k)
  __shared__ float ssum[16], ssq[16];
  const int tid = threadIdx.x, lane = tid & 31, wave = tid >> 5;
  const int m0 = blockIdx.x * 16;
  const int h = lane >> 4, nn = lane & 15;

  if (tid < 16) { ssum[tid] = 0.f; ssq[tid] = 0.f; }

  v8f acc[2][4];
#pragma unroll
  for (int i = 0; i < 2; ++i)
#pragma unroll
    for (int j = 0; j < 4; ++j) acc[i][j] = zero8();

  for (int k0 = 0; k0 < 1024; k0 += 32) {
#pragma unroll
    for (int half = 0; half < 2; ++half) {
      __syncthreads();                         // prev compute done before restage
      if (wave == 0) {
        if (half == 0)
          tdm_load_2d(lds_off(As), ao + (size_t)m0 * 1024 + k0, 32, 16, 1024, 3, 3);
        tdm_load_2d(lds_off(Bs), wpT + (size_t)(half * 512) * 1024 + k0,
                    32, 512, 1024, 3, 3);
        __builtin_amdgcn_s_wait_tensorcnt(0);
      }
      __syncthreads();
      v16bf a = load_afrag(As, 40, lane);
#pragma unroll
      for (int nt = 0; nt < 4; ++nt) {
        v16bf bb = load_bfragT(Bs + (wave * 64 + nt * 16) * 40, 40, lane);
        acc[half][nt] = wmma_bf16(a, bb, acc[half][nt]);
      }
    }
  }
  __syncthreads();

  // h = x + proj + bias; accumulate per-row sums for LN.
  float ps[8], ps2[8];
#pragma unroll
  for (int r = 0; r < 8; ++r) { ps[r] = 0.f; ps2[r] = 0.f; }
#pragma unroll
  for (int half = 0; half < 2; ++half)
#pragma unroll
    for (int nt = 0; nt < 4; ++nt)
#pragma unroll
      for (int r = 0; r < 8; ++r) {
        const int col = half * 512 + wave * 64 + nt * 16 + nn;
        const int row = r + 8 * h;
        const float hv = acc[half][nt][r] + bproj[col] +
                         x[(size_t)(m0 + row) * 1024 + col];
        acc[half][nt][r] = hv;
        ps[r] += hv; ps2[r] += hv * hv;
      }
#pragma unroll
  for (int r = 0; r < 8; ++r) {
    atomicAdd(&ssum[r + 8 * h], ps[r]);        // ds_add_f32
    atomicAdd(&ssq[r + 8 * h], ps2[r]);
  }
  __syncthreads();
#pragma unroll
  for (int half = 0; half < 2; ++half)
#pragma unroll
    for (int nt = 0; nt < 4; ++nt)
#pragma unroll
      for (int r = 0; r < 8; ++r) {
        const int col = half * 512 + wave * 64 + nt * 16 + nn;
        const int row = r + 8 * h;
        const float mu  = ssum[row] * (1.0f / 1024.0f);
        const float var = ssq[row] * (1.0f / 1024.0f) - mu * mu;
        const float o = gamma[col] * (acc[half][nt][r] - mu) * rsqrtf(var + 1e-5f) +
                        beta[col];
        out[(size_t)(m0 + row) * 1024 + col] = o;
      }
}

// ------------------------------------------------------------------ launch ---

extern "C" void kernel_launch(void* const* d_in, const int* in_sizes, int n_in,
                              void* d_out, int out_size, void* d_ws, size_t ws_size,
                              hipStream_t stream) {
  (void)in_sizes; (void)n_in; (void)out_size; (void)ws_size;
  const float* x     = (const float*)d_in[0];
  const float* Wqkv  = (const float*)d_in[1];
  const float* Wproj = (const float*)d_in[2];
  const float* bproj = (const float*)d_in[3];
  const float* gamma = (const float*)d_in[4];
  const float* beta  = (const float*)d_in[5];
  float* out = (float*)d_out;

  const size_t XE = (size_t)8 * 4096 * 1024;       // 33,554,432 elems
  char* ws = (char*)d_ws;
  size_t o = 0;
  bf16* xb  = (bf16*)(ws + o); o += XE * 2;
  bf16* wqT = (bf16*)(ws + o); o += (size_t)1024 * 3072 * 2;   // WqkvT [3072][1024]
  bf16* wpT = (bf16*)(ws + o); o += (size_t)1024 * 1024 * 2;   // WprojT [1024][1024]
  bf16* qb  = (bf16*)(ws + o); o += XE * 2;
  bf16* kb  = (bf16*)(ws + o); o += XE * 2;
  bf16* vb  = (bf16*)(ws + o); o += XE * 2;
  bf16* ao  = (bf16*)(ws + o); o += XE * 2;        // ~344 MB total

  cvt_f32_bf16<<<(int)(XE / 4 / 256), 256, 0, stream>>>(x, xb, (int)(XE / 4));
  cvt_f32_bf16_t<<<dim3(3072 / 32, 1024 / 32), 256, 0, stream>>>(Wqkv, wqT, 1024, 3072);
  cvt_f32_bf16_t<<<dim3(1024 / 32, 1024 / 32), 256, 0, stream>>>(Wproj, wpT, 1024, 1024);

  qkv_gemm_kernel<<<dim3(32768 / 128, 3072 / 128), 256, 0, stream>>>(xb, wqT, qb, kb, vb);
  chan_attn_kernel<<<128, 256, 0, stream>>>(qb, kb, vb, ao);
  proj_ln_kernel<<<32768 / 16, 256, 0, stream>>>(ao, wpT, x, bproj, gamma, beta, out);
}
